// PRMPConv_1099511628110
// MI455X (gfx1250) — matrix operand order repaired and checked by reading, not compiled, not run on hardware
//
#include <hip/hip_runtime.h>
#include <hip/hip_bf16.h>

typedef __bf16 bf16;
typedef __attribute__((ext_vector_type(8)))  __bf16 v8bf;
typedef __attribute__((ext_vector_type(16))) __bf16 v16bf;
typedef __attribute__((ext_vector_type(8)))  float  v8f;

#define DIM 128

// ---------------------------------------------------------------------------
// Prep: transpose weights to N-major bf16 so WMMA B-fragments are contiguous.
// W1,W2: [128][128] row-major (k-major).  Wu: [256][128].
// ---------------------------------------------------------------------------
__global__ void prep_weights(const float* __restrict__ W1,
                             const float* __restrict__ W2,
                             const float* __restrict__ Wu,
                             bf16* __restrict__ W1t,   // [n=128][k=128]
                             bf16* __restrict__ W2t,   // [n=128][k=128]
                             bf16* __restrict__ Wut) { // [n=128][k=256]
  int tid = blockIdx.x * blockDim.x + threadIdx.x;   // 0..32767
  if (tid < 128 * 128) {
    int n = tid & 127, k = tid >> 7;
    W1t[n * 128 + k] = (bf16)W1[k * 128 + n];
    W2t[n * 128 + k] = (bf16)W2[k * 128 + n];
  }
  if (tid < 256 * 128) {
    int n = tid & 127, k = tid >> 7;                 // k in 0..255
    Wut[n * 256 + k] = (bf16)Wu[k * 128 + n];
  }
}

// ---------------------------------------------------------------------------
// Convert x_dst to bf16 (GEMM1 A-matrix) and into cols 0..127 of concat buf.
// ---------------------------------------------------------------------------
__global__ void convert_xdst(const float* __restrict__ x_dst,
                             bf16* __restrict__ Xd,     // [N][128]
                             bf16* __restrict__ Xcat,   // [N][256]
                             int total) {
  int i = blockIdx.x * blockDim.x + threadIdx.x;
  if (i >= total) return;
  int node = i >> 7, c = i & 127;
  bf16 v = (bf16)x_dst[i];
  Xd[i] = v;
  Xcat[(size_t)node * 256 + c] = v;
}

// ---------------------------------------------------------------------------
// WMMA GEMM: C[M x 128] = A[M x K](bf16,row-major) * Bt[128 x K](bf16,N-major)
// + bias.  Block = 256 threads = 8 waves; block owns one 16-row strip, wave w
// owns N-tile n0 = 16w.  A strip staged once through LDS, shared by 8 waves.
// mode==1: relu + store bf16 (outBF);  mode==0: store f32 (outF).
// ---------------------------------------------------------------------------
__global__ __launch_bounds__(256) void gemm_wmma(
    const bf16* __restrict__ A, const bf16* __restrict__ Bt,
    const float* __restrict__ bias, float* __restrict__ outF,
    bf16* __restrict__ outBF, int K, int mode) {
  __shared__ __align__(16) bf16 tileA[16 * 256];     // up to K=256 -> 8 KB

  int tid  = threadIdx.x;
  int m0   = blockIdx.x * 16;

  // cooperative contiguous copy: rows m0..m0+15 of full width K
  {
    const uint4* g = (const uint4*)(A + (size_t)m0 * K);
    uint4* l = (uint4*)tileA;
    int nchunks = (16 * K) >> 3;                     // 16B chunks
    for (int i = tid; i < nchunks; i += 256) l[i] = g[i];
  }
  __syncthreads();

  int lane = tid & 31;
  int w    = tid >> 5;
  int nl   = lane & 15;
  int half = lane >> 4;
  int n0   = w * 16;

  v8f c = {0.f, 0.f, 0.f, 0.f, 0.f, 0.f, 0.f, 0.f};
  const bf16* arow = tileA + nl * K;
  const bf16* brow = Bt + (size_t)(n0 + nl) * K;

  for (int kt = 0; kt < K; kt += 32) {
    // A frag: lane l<16 -> row l, K = kt+{0..7, 16..23}; l>=16 -> K +8
    v8bf alo = *(const v8bf*)(arow + kt + half * 8);
    v8bf ahi = *(const v8bf*)(arow + kt + 16 + half * 8);
    v16bf a = __builtin_shufflevector(alo, ahi, 0, 1, 2, 3, 4, 5, 6, 7,
                                      8, 9, 10, 11, 12, 13, 14, 15);
    // B frag: lane l holds column n0+l, K = kt+{0..7, 16..23}; l>=16 -> K +8
    v8bf blo = *(const v8bf*)(brow + kt + half * 8);
    v8bf bhi = *(const v8bf*)(brow + kt + 16 + half * 8);
    v16bf b = __builtin_shufflevector(blo, bhi, 0, 1, 2, 3, 4, 5, 6, 7,
                                      8, 9, 10, 11, 12, 13, 14, 15);
    c = __builtin_amdgcn_wmma_f32_16x16x32_bf16(
        false, a, false, b, (short)0, c, false, false);
  }

  int col = n0 + nl;
  float bv = bias[col];
#pragma unroll
  for (int i = 0; i < 8; ++i) {
    int row = m0 + i + half * 8;                     // C layout (ISA 7.12.2)
    float v = c[i] + bv;
    if (mode) {
      v = fmaxf(v, 0.f);
      outBF[(size_t)row * DIM + col] = (bf16)v;
    } else {
      outF[(size_t)row * DIM + col] = v;
    }
  }
}

// ---------------------------------------------------------------------------
// Per-edge: d = x_src[src] - P[dst]; LayerNorm(d); atomic scatter into aggr.
// One wave32 per edge, 4 channels per lane.
// ---------------------------------------------------------------------------
__global__ __launch_bounds__(256) void edge_kernel(
    const float* __restrict__ x_src, const float* __restrict__ P,
    const long long* __restrict__ ei, const float* __restrict__ gamma,
    const float* __restrict__ beta, float* __restrict__ aggr,
    float* __restrict__ cnt, int E) {
  int e = (int)((blockIdx.x * (unsigned)blockDim.x + threadIdx.x) >> 5);
  if (e >= E) return;
  int lane = threadIdx.x & 31;
  int c0 = lane * 4;

  long long src = ei[e];
  long long dst = ei[(size_t)E + e];

  float4 xj = *(const float4*)(x_src + (size_t)src * DIM + c0);
  float4 pr = *(const float4*)(P + (size_t)dst * DIM + c0);
  float4 d;
  d.x = xj.x - pr.x; d.y = xj.y - pr.y; d.z = xj.z - pr.z; d.w = xj.w - pr.w;

  float s = d.x + d.y + d.z + d.w;
  float q = d.x * d.x + d.y * d.y + d.z * d.z + d.w * d.w;
#pragma unroll
  for (int m = 16; m >= 1; m >>= 1) {
    s += __shfl_xor(s, m, 32);
    q += __shfl_xor(q, m, 32);
  }
  float mu = s * (1.f / 128.f);
  float var = q * (1.f / 128.f) - mu * mu;
  float rstd = rsqrtf(var + 1e-5f);

  float4 g = *(const float4*)(gamma + c0);
  float4 b = *(const float4*)(beta + c0);
  float4 r;
  r.x = (d.x - mu) * rstd * g.x + b.x;
  r.y = (d.y - mu) * rstd * g.y + b.y;
  r.z = (d.z - mu) * rstd * g.z + b.z;
  r.w = (d.w - mu) * rstd * g.w + b.w;

  float* ap = aggr + (size_t)dst * DIM + c0;
  unsafeAtomicAdd(ap + 0, r.x);
  unsafeAtomicAdd(ap + 1, r.y);
  unsafeAtomicAdd(ap + 2, r.z);
  unsafeAtomicAdd(ap + 3, r.w);
  if (lane == 0) unsafeAtomicAdd(cnt + dst, 1.0f);
}

// ---------------------------------------------------------------------------
// aggr / max(cnt,1) -> bf16 into cols 128..255 of concat buffer.
// ---------------------------------------------------------------------------
__global__ void finalize_aggr(const float* __restrict__ aggr,
                              const float* __restrict__ cnt,
                              bf16* __restrict__ Xcat, int total) {
  int i = blockIdx.x * blockDim.x + threadIdx.x;
  if (i >= total) return;
  int node = i >> 7, c = i & 127;
  float denom = fmaxf(cnt[node], 1.0f);
  Xcat[(size_t)node * 256 + 128 + c] = (bf16)(aggr[i] / denom);
}

// ---------------------------------------------------------------------------
extern "C" void kernel_launch(void* const* d_in, const int* in_sizes, int n_in,
                              void* d_out, int out_size, void* d_ws,
                              size_t ws_size, hipStream_t stream) {
  const float*     x_src = (const float*)d_in[0];
  const float*     x_dst = (const float*)d_in[1];
  const long long* eidx  = (const long long*)d_in[2];   // int64 [2][E]
  const float*     W1    = (const float*)d_in[3];
  const float*     b1    = (const float*)d_in[4];
  const float*     W2    = (const float*)d_in[5];
  const float*     b2    = (const float*)d_in[6];
  const float*     gamma = (const float*)d_in[7];
  const float*     beta  = (const float*)d_in[8];
  const float*     Wu    = (const float*)d_in[9];
  const float*     bu    = (const float*)d_in[10];

  const int N_DST = in_sizes[1] / DIM;
  const int E     = in_sizes[2] / 2;

  // workspace layout
  char* ws = (char*)d_ws;
  auto al = [](size_t x) { return (x + 255) & ~(size_t)255; };
  size_t o = 0;
  bf16*  Xd    = (bf16*)(ws + o);  o += al((size_t)N_DST * DIM * 2);
  bf16*  W1t   = (bf16*)(ws + o);  o += al((size_t)128 * 128 * 2);
  bf16*  W2t   = (bf16*)(ws + o);  o += al((size_t)128 * 128 * 2);
  bf16*  Wut   = (bf16*)(ws + o);  o += al((size_t)128 * 256 * 2);
  bf16*  H1    = (bf16*)(ws + o);  o += al((size_t)N_DST * DIM * 2);
  float* P     = (float*)(ws + o); o += al((size_t)N_DST * DIM * 4);
  float* aggr  = (float*)(ws + o); o += al((size_t)N_DST * DIM * 4);
  float* cnt   = (float*)(ws + o); o += al((size_t)N_DST * 4);
  bf16*  Xcat  = (bf16*)(ws + o);  o += al((size_t)N_DST * 256 * 2);

  const int nd_elems = N_DST * DIM;
  const int m_tiles  = N_DST / 16;           // 50000/16 = 3125 exact

  prep_weights<<<(256 * 128 + 255) / 256, 256, 0, stream>>>(W1, W2, Wu, W1t,
                                                            W2t, Wut);
  convert_xdst<<<(nd_elems + 255) / 256, 256, 0, stream>>>(x_dst, Xd, Xcat,
                                                           nd_elems);
  // P = relu(Xd @ W1 + b1) @ W2 + b2   (node-level MLP hoisted off edges)
  gemm_wmma<<<m_tiles, 256, 0, stream>>>(Xd, W1t, b1, nullptr, H1, 128, 1);
  gemm_wmma<<<m_tiles, 256, 0, stream>>>(H1, W2t, b2, P, nullptr, 128, 0);

  hipMemsetAsync(aggr, 0, (size_t)nd_elems * 4, stream);
  hipMemsetAsync(cnt, 0, (size_t)N_DST * 4, stream);

  edge_kernel<<<(E + 7) / 8, 256, 0, stream>>>(x_src, P, eidx, gamma, beta,
                                               aggr, cnt, E);
  finalize_aggr<<<(nd_elems + 255) / 256, 256, 0, stream>>>(aggr, cnt, Xcat,
                                                            nd_elems);
  // out = concat(x_dst, aggr) @ Wu + bu
  gemm_wmma<<<m_tiles, 256, 0, stream>>>(Xcat, Wut, bu, (float*)d_out, nullptr,
                                         256, 0);
}
